// GraphTransformerAccidentModel_83425444757867
// MI455X (gfx1250) — matrix operand order repaired and checked by reading, not compiled, not run on hardware
//
#include <hip/hip_runtime.h>
#include <hip/hip_bf16.h>

typedef _Float16 v16h __attribute__((ext_vector_type(16)));
typedef float    v8f  __attribute__((ext_vector_type(8)));
typedef _Float16 h2v  __attribute__((ext_vector_type(2)));
typedef float    f2v  __attribute__((ext_vector_type(2)));

#define KF    4096      // GEMM K for layer1 (feature part of 4097)
#define D_IN  4097
#define DH1   512
#define DH2   256
#define DFF   512
#define NB    2
#define TT    100
#define NN    30
#define ROWS  6000      // B*T*N
#define BT    200       // B*T
#define INV29 (1.0f/29.0f)

// ---------------------------------------------------------------------------
// Prep: swizzle combined layer-1 weight (W1a - W2a/29) over K=0..4095 into the
// WMMA B-operand layout as f16.  Block layout: [kk(128)][nt(32)][lane(32)][e(16)]
// where k = kk*32 + (lane>>4)*16 + e, n = nt*16 + (lane&15).
// ---------------------------------------------------------------------------
__global__ __launch_bounds__(256)
void prep_w1_kernel(const float* __restrict__ W1a, const float* __restrict__ W2a,
                    _Float16* __restrict__ bsw) {
    int idx  = blockIdx.x * 256 + threadIdx.x;          // < 4096*512 = 2097152
    int e    = idx & 15;
    int lane = (idx >> 4) & 31;
    int blk  = idx >> 9;                                // kk*32 + nt
    int nt   = blk & 31;
    int kk   = blk >> 5;
    int k    = kk * 32 + (lane >> 4) * 16 + e;
    int n    = nt * 16 + (lane & 15);
    float v  = W1a[(size_t)k * DH1 + n] - W2a[(size_t)k * DH1 + n] * INV29;
    bsw[idx] = (_Float16)v;
}

// Same for layer-2 combined weight (W1b - W2b/29), K=512, N=256.
__global__ __launch_bounds__(256)
void prep_w2_kernel(const float* __restrict__ W1b, const float* __restrict__ W2b,
                    _Float16* __restrict__ bsw) {
    int idx  = blockIdx.x * 256 + threadIdx.x;          // < 512*256 = 131072
    int e    = idx & 15;
    int lane = (idx >> 4) & 31;
    int blk  = idx >> 9;                                // kk*16 + nt
    int nt   = blk & 15;
    int kk   = blk >> 4;
    int k    = kk * 32 + (lane >> 4) * 16 + e;
    int n    = nt * 16 + (lane & 15);
    float v  = W1b[(size_t)k * DH2 + n] - W2b[(size_t)k * DH2 + n] * INV29;
    bsw[idx] = (_Float16)v;
}

// Depth-row (K=4096) of the combined layer-1 weight, fp32.
__global__ __launch_bounds__(256)
void prep_dw_kernel(const float* __restrict__ W1a, const float* __restrict__ W2a,
                    float* __restrict__ dw) {
    int n = blockIdx.x * 256 + threadIdx.x;
    if (n < DH1)
        dw[n] = W1a[(size_t)KF * DH1 + n] - W2a[(size_t)KF * DH1 + n] * INV29;
}

// ---------------------------------------------------------------------------
// St1[k][bt] = sum over 30 nodes of x[bt][node][k]   (transposed for coalesced
// reads in smallgemm).  k = 4096 is the depth column.
// ---------------------------------------------------------------------------
__global__ __launch_bounds__(256)
void sfeat_kernel(const float* __restrict__ feat, const float* __restrict__ depth,
                  float* __restrict__ St) {
    int bt = blockIdx.x;
    for (int k = threadIdx.x; k < D_IN; k += 256) {
        float s = 0.f;
        if (k < KF) {
            const float* p = feat + (size_t)(bt * NN) * KF + k;
            for (int j = 0; j < NN; ++j) s += p[(size_t)j * KF];
        } else {
            const float* p = depth + bt * NN;
            for (int j = 0; j < NN; ++j) s += p[j];
        }
        St[(size_t)k * BT + bt] = s;
    }
}

// St2[k][bt] = sum over 30 nodes of h[bt][node][k], k < 512.
__global__ __launch_bounds__(256)
void sh_kernel(const _Float16* __restrict__ h, float* __restrict__ St) {
    int bt = blockIdx.x;
    for (int k = threadIdx.x; k < DH1; k += 256) {
        float s = 0.f;
        const _Float16* p = h + (size_t)(bt * NN) * DH1 + k;
        for (int j = 0; j < NN; ++j) s += (float)p[(size_t)j * DH1];
        St[(size_t)k * BT + bt] = s;
    }
}

// ---------------------------------------------------------------------------
// Generic neighbor-term GEMM:  out[bt][c] = bias[c] + (1/29)*sum_k St[k][bt]*W[k][c]
// grid.x = N/16.  W chunk (256 x 16) staged in LDS and read exactly once per
// block (total W traffic = |W|, vs 200x|W| for the naive version).
// ---------------------------------------------------------------------------
__global__ __launch_bounds__(256)
void smallgemm_kernel(const float* __restrict__ St, const float* __restrict__ W,
                      const float* __restrict__ bias, float* __restrict__ outp,
                      int K, int N) {
    __shared__ float Wl[256 * 16];
    int c0 = blockIdx.x * 16;
    int bt = threadIdx.x;
    float acc[16];
#pragma unroll
    for (int j = 0; j < 16; ++j) acc[j] = 0.f;
    for (int k0 = 0; k0 < K; k0 += 256) {
        int kcnt = (K - k0) < 256 ? (K - k0) : 256;
        __syncthreads();
        for (int idx = threadIdx.x; idx < kcnt * 16; idx += 256) {
            int kk = idx >> 4, j = idx & 15;
            Wl[idx] = W[(size_t)(k0 + kk) * N + c0 + j];
        }
        __syncthreads();
        if (bt < BT) {
            for (int kk = 0; kk < kcnt; ++kk) {
                float s = St[(size_t)(k0 + kk) * BT + bt];   // coalesced over bt
#pragma unroll
                for (int j = 0; j < 16; ++j) acc[j] += s * Wl[kk * 16 + j];  // LDS broadcast
            }
        }
    }
    if (bt < BT) {
#pragma unroll
        for (int j = 0; j < 16; ++j)
            outp[bt * N + c0 + j] = acc[j] * INV29 + bias[c0 + j];
    }
}

// ---------------------------------------------------------------------------
// Layer-1 GEMM: h = relu( feat @ Wself1[0:4096] + depth*dw1 + mvec1 )
// M=6000 (pad to 6016), K=4096, N=512.
// Each wave: 32(M) x 128(N) block = 2 M-tiles x 8 N-tiles, 16 f32 accumulators.
// Each B fragment feeds two WMMAs (A0,A1) -> vmem:wmma = 24:16 per k-step.
// ---------------------------------------------------------------------------
__global__ __launch_bounds__(128)
void gemm1_kernel(const float* __restrict__ feat, const float* __restrict__ depth,
                  const _Float16* __restrict__ bsw, const float* __restrict__ mv1,
                  const float* __restrict__ dw1, _Float16* __restrict__ hout) {
    int wv   = blockIdx.x * 4 + (threadIdx.x >> 5);   // 0..751
    int lane = threadIdx.x & 31;
    int mp   = wv >> 2;                                // 0..187 (32-row block)
    int ns   = wv & 3;                                 // 0..3  (128-col strip)
    int half = lane >> 4;
    int r0   = mp * 32 + (lane & 15);
    int r1   = r0 + 16;
    const float* arow0 = feat + (size_t)r0 * KF;
    const float* arow1 = feat + (size_t)(r1 < ROWS ? r1 : ROWS - 1) * KF;

    v8f acc[16];
#pragma unroll
    for (int t = 0; t < 16; ++t)
#pragma unroll
        for (int e = 0; e < 8; ++e) acc[t][e] = 0.f;

    for (int kk = 0; kk < 128; ++kk) {
        int k0 = kk * 32;
        v16h a0, a1;
#pragma unroll
        for (int j = 0; j < 8; ++j) {
            // A layout: VGPR j -> K = k0 + 16*half + 2*j .. +1
            int koff = k0 + (half << 4) + (j << 1);
            f2v f0 = *(const f2v*)(arow0 + koff);
            f2v f1 = *(const f2v*)(arow1 + koff);
            a0[2 * j] = (_Float16)f0.x;  a0[2 * j + 1] = (_Float16)f0.y;
            a1[2 * j] = (_Float16)f1.x;  a1[2 * j + 1] = (_Float16)f1.y;
        }
#pragma unroll
        for (int t = 0; t < 8; ++t) {
            int nt = ns * 8 + t;
            v16h b = *(const v16h*)(bsw + (((size_t)kk * 32 + nt) * 32 + lane) * 16);
            acc[t]     = __builtin_amdgcn_wmma_f32_16x16x32_f16(
                false, a0, false, b, (short)0, acc[t],     false, false);
            acc[8 + t] = __builtin_amdgcn_wmma_f32_16x16x32_f16(
                false, a1, false, b, (short)0, acc[8 + t], false, false);
        }
    }
    // Epilogue: + depth*dw1[n] + mvec1[bt][n], ReLU, store f16 (guard M pad).
#pragma unroll
    for (int g = 0; g < 2; ++g) {
#pragma unroll
        for (int t = 0; t < 8; ++t) {
            int ncol = (ns * 8 + t) * 16 + (lane & 15);
            float dw = dw1[ncol];
#pragma unroll
            for (int rr = 0; rr < 8; ++rr) {
                int r2 = mp * 32 + g * 16 + half * 8 + rr;   // C/D row mapping
                if (r2 < ROWS) {
                    int bt = r2 / NN;
                    float v = acc[g * 8 + t][rr] + depth[r2] * dw + mv1[bt * DH1 + ncol];
                    v = v > 0.f ? v : 0.f;
                    hout[(size_t)r2 * DH1 + ncol] = (_Float16)v;
                }
            }
        }
    }
}

// ---------------------------------------------------------------------------
// Layer-2 GEMM: h2 = relu( h @ Wself2 + mvec2 ).  M=6000, K=512, N=256.
// ---------------------------------------------------------------------------
__global__ __launch_bounds__(128)
void gemm2_kernel(const _Float16* __restrict__ h, const _Float16* __restrict__ bsw,
                  const float* __restrict__ mv2, float* __restrict__ h2out) {
    int wv   = blockIdx.x * 4 + (threadIdx.x >> 5);   // 0..1499
    int lane = threadIdx.x & 31;
    int mt   = wv >> 2;                                // 0..374
    int ns   = wv & 3;                                 // 0..3
    int half = lane >> 4;
    const _Float16* arow = h + (size_t)(mt * 16 + (lane & 15)) * DH1;

    v8f acc[4];
#pragma unroll
    for (int t = 0; t < 4; ++t)
#pragma unroll
        for (int e = 0; e < 8; ++e) acc[t][e] = 0.f;

    for (int kk = 0; kk < 16; ++kk) {
        int k0 = kk * 32;
        v16h a;
#pragma unroll
        for (int j = 0; j < 8; ++j) {
            int koff = k0 + (half << 4) + (j << 1);
            h2v p = *(const h2v*)(arow + koff);
            a[2 * j]     = p.x;
            a[2 * j + 1] = p.y;
        }
#pragma unroll
        for (int t = 0; t < 4; ++t) {
            int nt = ns * 4 + t;
            v16h b = *(const v16h*)(bsw + (((size_t)kk * 16 + nt) * 32 + lane) * 16);
            acc[t] = __builtin_amdgcn_wmma_f32_16x16x32_f16(
                false, a, false, b, (short)0, acc[t], false, false);
        }
    }
#pragma unroll
    for (int t = 0; t < 4; ++t) {
        int ncol = (ns * 4 + t) * 16 + (lane & 15);
#pragma unroll
        for (int rr = 0; rr < 8; ++rr) {
            int r2 = mt * 16 + half * 8 + rr;
            int bt = r2 / NN;
            float v = acc[t][rr] + mv2[bt * DH2 + ncol];
            h2out[(size_t)r2 * DH2 + ncol] = v > 0.f ? v : 0.f;
        }
    }
}

// g[bt][c] = mean over 30 nodes of h2
__global__ __launch_bounds__(256)
void gpool_kernel(const float* __restrict__ h2, float* __restrict__ g) {
    int bt = blockIdx.x, c = threadIdx.x;
    float s = 0.f;
    const float* p = h2 + (size_t)(bt * NN) * DH2 + c;
    for (int j = 0; j < NN; ++j) s += p[j * DH2];
    g[bt * DH2 + c] = s * (1.0f / 30.0f);
}

// q/k/v = g @ W{q,k,v}  (blockIdx.y selects which)
__global__ __launch_bounds__(256)
void qkv_kernel(const float* __restrict__ g, const float* __restrict__ Wq,
                const float* __restrict__ Wk, const float* __restrict__ Wv,
                float* __restrict__ qkv) {
    __shared__ float row[DH2];
    int r = blockIdx.x, which = blockIdx.y, c = threadIdx.x;
    row[c] = g[r * DH2 + c];
    __syncthreads();
    const float* W = (which == 0) ? Wq : (which == 1) ? Wk : Wv;
    float acc = 0.f;
    for (int k = 0; k < DH2; ++k) acc += row[k] * W[k * DH2 + c];
    qkv[((size_t)which * BT + r) * DH2 + c] = acc;
}

// Per (b,head) attention with streaming softmax.  K/V rows live in LDS.
__global__ __launch_bounds__(128)
void attn_kernel(const float* __restrict__ qkv, float* __restrict__ ctx) {
    __shared__ float kb[TT * 64];
    __shared__ float vb[TT * 64];
    int b  = blockIdx.x >> 2;
    int hh = blockIdx.x & 3;
    const float* qws = qkv;
    const float* kws = qkv + (size_t)BT * DH2;
    const float* vws = qkv + (size_t)2 * BT * DH2;
    for (int idx = threadIdx.x; idx < TT * 64; idx += 128) {
        int t = idx >> 6, d = idx & 63;
        kb[idx] = kws[(size_t)(b * TT + t) * DH2 + hh * 64 + d];
        vb[idx] = vws[(size_t)(b * TT + t) * DH2 + hh * 64 + d];
    }
    __syncthreads();
    int t = threadIdx.x;
    if (t < TT) {
        float qr[64];
        const float* qp = qws + (size_t)(b * TT + t) * DH2 + hh * 64;
#pragma unroll
        for (int d = 0; d < 64; ++d) qr[d] = qp[d];
        float mx = -1e30f;
        for (int s = 0; s < TT; ++s) {
            float sc = 0.f;
#pragma unroll
            for (int d = 0; d < 64; ++d) sc += qr[d] * kb[s * 64 + d];
            sc *= 0.125f;                       // 1/sqrt(64)
            mx = fmaxf(mx, sc);
        }
        float cx[64];
#pragma unroll
        for (int d = 0; d < 64; ++d) cx[d] = 0.f;
        float den = 0.f;
        for (int s = 0; s < TT; ++s) {
            float sc = 0.f;
#pragma unroll
            for (int d = 0; d < 64; ++d) sc += qr[d] * kb[s * 64 + d];
            float e = expf(sc * 0.125f - mx);
            den += e;
#pragma unroll
            for (int d = 0; d < 64; ++d) cx[d] += e * vb[s * 64 + d];
        }
        float inv = 1.f / den;
        float* cp = ctx + (size_t)(b * TT + t) * DH2 + hh * 64;
#pragma unroll
        for (int d = 0; d < 64; ++d) cp[d] = cx[d] * inv;
    }
}

// y = LN1( g + ctx @ Wo )
__global__ __launch_bounds__(256)
void oln_kernel(const float* __restrict__ ctx, const float* __restrict__ Wo,
                const float* __restrict__ g, const float* __restrict__ ln1g,
                const float* __restrict__ ln1b, float* __restrict__ y) {
    __shared__ float crow[DH2];
    __shared__ float red[256];
    int r = blockIdx.x, c = threadIdx.x;
    crow[c] = ctx[r * DH2 + c];
    __syncthreads();
    float acc = 0.f;
    for (int k = 0; k < DH2; ++k) acc += crow[k] * Wo[k * DH2 + c];
    float s = g[r * DH2 + c] + acc;
    red[c] = s; __syncthreads();
    for (int st = 128; st > 0; st >>= 1) { if (c < st) red[c] += red[c + st]; __syncthreads(); }
    float mu = red[0] * (1.f / DH2); __syncthreads();
    float d = s - mu;
    red[c] = d * d; __syncthreads();
    for (int st = 128; st > 0; st >>= 1) { if (c < st) red[c] += red[c + st]; __syncthreads(); }
    float var = red[0] * (1.f / DH2);
    y[r * DH2 + c] = d * rsqrtf(var + 1e-5f) * ln1g[c] + ln1b[c];
}

// ff = relu( y @ Wf1 + bf1 )
__global__ __launch_bounds__(256)
void ff1_kernel(const float* __restrict__ y, const float* __restrict__ Wf1,
                const float* __restrict__ bf1, float* __restrict__ ff) {
    __shared__ float row[DH2];
    int r = blockIdx.x, c = threadIdx.x;
    row[c] = y[r * DH2 + c];
    __syncthreads();
    for (int cc = c; cc < DFF; cc += 256) {
        float acc = bf1[cc];
        for (int k = 0; k < DH2; ++k) acc += row[k] * Wf1[k * DFF + cc];
        ff[(size_t)r * DFF + cc] = fmaxf(acc, 0.f);
    }
}

// z = LN2( y + ff @ Wf2 + bf2 );  out = sigmoid(z @ Wc + bc);  uncertainty = 0
__global__ __launch_bounds__(256)
void final_kernel(const float* __restrict__ y, const float* __restrict__ ff,
                  const float* __restrict__ Wf2, const float* __restrict__ bf2,
                  const float* __restrict__ ln2g, const float* __restrict__ ln2b,
                  const float* __restrict__ Wc, const float* __restrict__ bc,
                  float* __restrict__ out) {
    __shared__ float fr[DFF];
    __shared__ float red[256];
    int r = blockIdx.x, c = threadIdx.x;
    fr[c]       = ff[(size_t)r * DFF + c];
    fr[c + 256] = ff[(size_t)r * DFF + c + 256];
    __syncthreads();
    float acc = bf2[c];
    for (int k = 0; k < DFF; ++k) acc += fr[k] * Wf2[k * DH2 + c];
    float s = y[r * DH2 + c] + acc;
    red[c] = s; __syncthreads();
    for (int st = 128; st > 0; st >>= 1) { if (c < st) red[c] += red[c + st]; __syncthreads(); }
    float mu = red[0] * (1.f / DH2); __syncthreads();
    float d = s - mu;
    red[c] = d * d; __syncthreads();
    for (int st = 128; st > 0; st >>= 1) { if (c < st) red[c] += red[c + st]; __syncthreads(); }
    float var = red[0] * (1.f / DH2); __syncthreads();
    float z = d * rsqrtf(var + 1e-5f) * ln2g[c] + ln2b[c];
    red[c] = z * Wc[c]; __syncthreads();
    for (int st = 128; st > 0; st >>= 1) { if (c < st) red[c] += red[c + st]; __syncthreads(); }
    if (c == 0) {
        float logit = red[0] + bc[0];
        out[r]      = 1.f / (1.f + expf(-logit));   // probs
        out[BT + r] = 0.f;                          // uncertainty (eval mode)
    }
}

extern "C" void kernel_launch(void* const* d_in, const int* in_sizes, int n_in,
                              void* d_out, int out_size, void* d_ws, size_t ws_size,
                              hipStream_t stream) {
    const float* feat  = (const float*)d_in[0];
    const float* depth = (const float*)d_in[1];
    const float* W1a   = (const float*)d_in[2];
    const float* W2a   = (const float*)d_in[3];
    const float* b1a   = (const float*)d_in[4];
    const float* W1b   = (const float*)d_in[5];
    const float* W2b   = (const float*)d_in[6];
    const float* b1b   = (const float*)d_in[7];
    const float* Wq    = (const float*)d_in[8];
    const float* Wk    = (const float*)d_in[9];
    const float* Wv    = (const float*)d_in[10];
    const float* Wo    = (const float*)d_in[11];
    const float* ln1g  = (const float*)d_in[12];
    const float* ln1b  = (const float*)d_in[13];
    const float* Wf1   = (const float*)d_in[14];
    const float* bf1   = (const float*)d_in[15];
    const float* Wf2   = (const float*)d_in[16];
    const float* bf2   = (const float*)d_in[17];
    const float* ln2g  = (const float*)d_in[18];
    const float* ln2b  = (const float*)d_in[19];
    const float* Wc    = (const float*)d_in[20];
    const float* bc    = (const float*)d_in[21];
    float* out = (float*)d_out;

    char* ws = (char*)d_ws;
    size_t off = 0;
    auto walloc = [&](size_t bytes) -> char* {
        char* p = ws + off;
        off = (off + bytes + 255) & ~(size_t)255;
        return p;
    };
    _Float16* bsw1 = (_Float16*)walloc((size_t)KF  * DH1 * sizeof(_Float16)); // 4 MB
    _Float16* bsw2 = (_Float16*)walloc((size_t)DH1 * DH2 * sizeof(_Float16)); // 256 KB
    float*    dw1  = (float*)   walloc((size_t)DH1 * sizeof(float));
    float*    St1  = (float*)   walloc((size_t)D_IN * BT * sizeof(float));    // 3.3 MB
    float*    St2  = (float*)   walloc((size_t)DH1 * BT * sizeof(float));
    float*    mv1  = (float*)   walloc((size_t)BT  * DH1 * sizeof(float));
    _Float16* h    = (_Float16*)walloc((size_t)ROWS * DH1 * sizeof(_Float16)); // 6 MB
    float*    mv2  = (float*)   walloc((size_t)BT  * DH2 * sizeof(float));
    float*    h2   = (float*)   walloc((size_t)ROWS * DH2 * sizeof(float));    // 6 MB
    float*    g    = (float*)   walloc((size_t)BT  * DH2 * sizeof(float));
    float*    qkv  = (float*)   walloc((size_t)3 * BT * DH2 * sizeof(float));
    float*    ctx  = (float*)   walloc((size_t)BT  * DH2 * sizeof(float));
    float*    y    = (float*)   walloc((size_t)BT  * DH2 * sizeof(float));
    float*    ff   = (float*)   walloc((size_t)BT  * DFF * sizeof(float));

    prep_w1_kernel<<<(KF * DH1) / 256, 256, 0, stream>>>(W1a, W2a, bsw1);
    prep_w2_kernel<<<(DH1 * DH2) / 256, 256, 0, stream>>>(W1b, W2b, bsw2);
    prep_dw_kernel<<<2, 256, 0, stream>>>(W1a, W2a, dw1);
    sfeat_kernel<<<BT, 256, 0, stream>>>(feat, depth, St1);
    smallgemm_kernel<<<DH1 / 16, 256, 0, stream>>>(St1, W2a, b1a, mv1, D_IN, DH1);
    gemm1_kernel<<<188, 128, 0, stream>>>(feat, depth, bsw1, mv1, dw1, h);
    sh_kernel<<<BT, 256, 0, stream>>>(h, St2);
    smallgemm_kernel<<<DH2 / 16, 256, 0, stream>>>(St2, W2b, b1b, mv2, DH1, DH2);
    gemm2_kernel<<<375, 128, 0, stream>>>(h, bsw2, mv2, h2);
    gpool_kernel<<<BT, 256, 0, stream>>>(h2, g);
    qkv_kernel<<<dim3(BT, 3), 256, 0, stream>>>(g, Wq, Wk, Wv, qkv);
    attn_kernel<<<NB * 4, 128, 0, stream>>>(qkv, ctx);
    oln_kernel<<<BT, 256, 0, stream>>>(ctx, Wo, g, ln1g, ln1b, y);
    ff1_kernel<<<BT, 256, 0, stream>>>(y, Wf1, bf1, ff);
    final_kernel<<<BT, 256, 0, stream>>>(y, ff, Wf2, bf2, ln2g, ln2b, Wc, bc, out);
    (void)in_sizes; (void)n_in; (void)out_size; (void)ws_size;
}
